// GNNLSTMtest_82480551952453
// MI455X (gfx1250) — matrix operand order repaired and checked by reading, not compiled, not run on hardware
//
#include <hip/hip_runtime.h>
#include <hip/hip_bf16.h>
#include <cstdint>

typedef __attribute__((ext_vector_type(16))) _Float16 v16h;
typedef __attribute__((ext_vector_type(8)))  _Float16 v8h;
typedef __attribute__((ext_vector_type(8)))  float    v8f;

// Problem constants (from the reference)
constexpr int TS   = 60;      // timesteps
constexpr int NB   = 16384;   // new nodes (batch)
constexpr int NC   = 5;       // diagnosis codes
constexpr int HD   = 20;      // hidden/emb size
constexpr int GD   = 80;      // 4*H gates
constexpr int KCAT = 40;      // x(20) || h(20)
constexpr int N0C  = 3190;    // base nodes
constexpr int N0P  = 3200;    // padded to 16
constexpr int E0C  = 31900;   // base edges
constexpr int HCC  = 15;      // sage hidden channels
constexpr int NTOT = N0C + NB;
constexpr int NCLS = 3;
constexpr int NT0  = 200;     // 3200/16 similarity column tiles

// ---- WMMA f16 16x16x32 fragment K-maps (CDNA5 ISA §7.12.2) ----
// A (16x32, lane=row, hs=lane>>4): slots 0..7 -> k=hs*8+s ; slots 8..15 -> k=16+hs*8+(s-8)
// B (32x16, lane=col):             kmapB: contiguous k=hs*16+s (own order, not AMD's; valid only if A matches)
__device__ __forceinline__ int kmapA(int s, int hs) {
    return ((s < 8) ? 0 : 16) + hs * 8 + (s & 7);
}
__device__ __forceinline__ int kmapB(int s, int hs) { return hs * 16 + s; }

__device__ __forceinline__ float sigm(float x) { return 1.f / (1.f + __expf(-x)); }

__device__ __forceinline__ unsigned lds_off(const void* p) {
    // low 32 bits of a flat pointer into __shared__ == wave-relative LDS byte address
    return (unsigned)(uintptr_t)p;
}

// ============================================================================
// 1) Embedding: seq[t,n,h] = relu(sum_c x[t,n,c]*emb[c,h])  (f16 output)
// ============================================================================
__global__ __launch_bounds__(256)
void embed_kernel(const float* __restrict__ x, const float* __restrict__ emb,
                  _Float16* __restrict__ seq)
{
    int idx = blockIdx.x * 256 + threadIdx.x;
    if (idx >= TS * NB * HD) return;
    int h  = idx % HD;
    int tn = idx / HD;
    const float* xp = x + (size_t)tn * NC;
    float acc = 0.f;
#pragma unroll
    for (int c = 0; c < NC; ++c) acc += xp[c] * emb[c * HD + h];
    seq[idx] = (_Float16)fmaxf(acc, 0.f);
}

// ============================================================================
// 2) One LSTM layer. 1 wave / 16-row batch tile, sweeps t=0..59 with state in
//    LDS. x_t tiles are DMA'd with global_load_async_to_lds (double-buffered);
//    A fragments are contiguous ds_load_b128 pairs from the unified
//    [x(0:20) | h(20:40) | 0(40:64)] LDS rows. Gate GEMM = 5 N-tiles x 2
//    K-chunks of v_wmma_f32_16x16x32_f16; weights stay in VGPRs all 60 steps.
// ============================================================================
__global__ __launch_bounds__(32)
void lstm_layer_kernel(const _Float16* __restrict__ in_seq,
                       _Float16* __restrict__ out_seq,
                       const float* __restrict__ Wih,   // [80,20]
                       const float* __restrict__ Whh,   // [80,20]
                       const float* __restrict__ bih,   // [80]
                       const float* __restrict__ bhh)   // [80]
{
    __shared__ __align__(16) _Float16 xcatS[2][16][64]; // rows: x|h|pad
    __shared__ float cS[16][HD];
    __shared__ float gS[16][GD];
    __shared__ float bS[GD];

    const int lane = threadIdx.x;
    const int col  = lane & 15;
    const int hs   = lane >> 4;
    const int m0   = blockIdx.x * 16;

    // zero h+pad columns of both buffers (x columns are DMA targets: untouched)
    {
        int bsel = lane >> 4, row = lane & 15;
        for (int j = HD; j < 64; ++j) xcatS[bsel][row][j] = (_Float16)0.f;
    }
    for (int i = lane; i < 16 * HD; i += 32) (&cS[0][0])[i] = 0.f;
    for (int g = lane; g < GD; g += 32) bS[g] = bih[g] + bhh[g];

    // B fragments: Wcat[k][g], k<20 -> Wih[g][k], 20<=k<40 -> Whh[g][k-20]
    v16h bf[5][2];
#pragma unroll
    for (int nt = 0; nt < 5; ++nt) {
#pragma unroll
        for (int kc = 0; kc < 2; ++kc) {
#pragma unroll
            for (int s = 0; s < 16; ++s) {
                int k = kc * 32 + kmapA(s, hs);
                int g = nt * 16 + col;
                float w = 0.f;
                if (k < HD)        w = Wih[g * HD + k];
                else if (k < KCAT) w = Whh[g * HD + (k - HD)];
                bf[nt][kc][s] = (_Float16)w;
            }
        }
    }

    const unsigned ldsX[2] = { lds_off(&xcatS[0][0][0]), lds_off(&xcatS[1][0][0]) };
    // async stage of one x-tile: 16 rows x 40B, b64 chunks (5 per row)
    auto issue_x = [&](int t, unsigned ldsbase) {
        uint64_t gbase = (uint64_t)(uintptr_t)(in_seq + ((size_t)t * NB + m0) * HD);
#pragma unroll
        for (int q = 0; q < 3; ++q) {
            int i = lane + 32 * q;
            if (i < 80) {
                unsigned r = (unsigned)(i / 5), cch = (unsigned)(i % 5);
                unsigned voff = r * 40u + cch * 8u;
                unsigned dst  = ldsbase + r * 128u + cch * 8u;
                asm volatile("global_load_async_to_lds_b64 %0, %1, %2"
                             :: "v"(dst), "v"(voff), "s"(gbase) : "memory");
            }
        }
    };
    issue_x(0, ldsX[0]);

    int buf = 0;
    for (int t = 0; t < TS; ++t) {
        asm volatile("s_wait_asynccnt 0" ::: "memory");
        __syncthreads();
        if (t + 1 < TS) issue_x(t + 1, ldsX[buf ^ 1]);

        // A fragments: contiguous 16B LDS loads per the layout map
        const _Float16* xr = &xcatS[buf][col][0];
        v8h a0lo = *(const v8h*)(xr + hs * 8);
        v8h a0hi = *(const v8h*)(xr + 16 + hs * 8);
        v8h a1lo = *(const v8h*)(xr + 32 + hs * 8);
        v8h a1hi = *(const v8h*)(xr + 48 + hs * 8);
        v16h a0 = __builtin_shufflevector(a0lo, a0hi, 0,1,2,3,4,5,6,7,8,9,10,11,12,13,14,15);
        v16h a1 = __builtin_shufflevector(a1lo, a1hi, 0,1,2,3,4,5,6,7,8,9,10,11,12,13,14,15);

#pragma unroll
        for (int nt = 0; nt < 5; ++nt) {
            v8f acc = {};
            acc = __builtin_amdgcn_wmma_f32_16x16x32_f16(false, a0, false, bf[nt][0],
                                                         (short)0, acc, false, false);
            acc = __builtin_amdgcn_wmma_f32_16x16x32_f16(false, a1, false, bf[nt][1],
                                                         (short)0, acc, false, false);
#pragma unroll
            for (int r = 0; r < 8; ++r)
                gS[hs * 8 + r][nt * 16 + col] = acc[r];
        }
        __syncthreads();

        // cell update: 320 cells, 10/lane. PyTorch gate order i,f,g,o.
        const int nbuf = buf ^ 1;
#pragma unroll
        for (int i = 0; i < 10; ++i) {
            int e = lane + 32 * i;
            int m = e / HD, j = e % HD;
            float ig = sigm (gS[m][j]          + bS[j]);
            float fg = sigm (gS[m][HD + j]     + bS[HD + j]);
            float gg = tanhf(gS[m][2 * HD + j] + bS[2 * HD + j]);
            float og = sigm (gS[m][3 * HD + j] + bS[3 * HD + j]);
            float c  = fg * cS[m][j] + ig * gg;
            _Float16 hh = (_Float16)(og * tanhf(c));
            cS[m][j] = c;
            xcatS[nbuf][m][HD + j] = hh;             // feeds step t+1
            out_seq[(size_t)t * NB * HD + (size_t)(m0 + m) * HD + j] = hh;
        }
        buf = nbuf;
    }
}

// ============================================================================
// 3) Self-attention over time: one block (64 threads) per node. Writes f32 xh
//    (for SAGE) and pre-normalized f16 xhn rows padded to 32 (for cosine WMMA).
// ============================================================================
__global__ __launch_bounds__(64)
void attn_kernel(const _Float16* __restrict__ seq,
                 const float* __restrict__ w1, const float* __restrict__ b1,
                 const float* __restrict__ w2, const float* __restrict__ b2,
                 float* __restrict__ xh, _Float16* __restrict__ xhn)
{
    __shared__ float w1S[64 * HD];
    __shared__ float hbuf[TS][HD];
    __shared__ float hw[64][TS];
    __shared__ float ew[TS];
    __shared__ float xsh[HD];
    __shared__ float sq[HD];
    __shared__ float invS;

    const int n = blockIdx.x;
    const int k = threadIdx.x;

    for (int i = k; i < 64 * HD; i += 64) w1S[i] = w1[i];
    for (int i = k; i < TS * HD; i += 64)
        (&hbuf[0][0])[i] =
            (float)seq[(size_t)(i / HD) * NB * HD + (size_t)n * HD + (i % HD)];
    __syncthreads();

    float w2k = w2[k];
    for (int t = 0; t < TS; ++t) {
        float acc = b1[k];
#pragma unroll
        for (int j = 0; j < HD; ++j) acc += w1S[k * HD + j] * hbuf[t][j];
        hw[k][t] = w2k * fmaxf(acc, 0.f);
    }
    __syncthreads();
    if (k < TS) {
        float e = b2[0];
        for (int q = 0; q < 64; ++q) e += hw[q][k];
        ew[k] = e;
    }
    __syncthreads();
    if (k == 0) {
        float mx = ew[0];
        for (int t = 1; t < TS; ++t) mx = fmaxf(mx, ew[t]);
        float s = 0.f;
        for (int t = 0; t < TS; ++t) { float v = __expf(ew[t] - mx); ew[t] = v; s += v; }
        float inv = 1.f / s;
        for (int t = 0; t < TS; ++t) ew[t] *= inv;
    }
    __syncthreads();
    if (k < HD) {
        float acc = 0.f;
        for (int t = 0; t < TS; ++t) acc += ew[t] * hbuf[t][k];
        xh[(size_t)n * HD + k] = acc;
        xsh[k] = acc;
        sq[k]  = acc * acc;
    }
    __syncthreads();
    if (k == 0) {
        float s = 0.f;
#pragma unroll
        for (int j = 0; j < HD; ++j) s += sq[j];
        invS = rsqrtf(fmaxf(s, 1e-30f));
    }
    __syncthreads();
    if (k < 32)
        xhn[(size_t)n * 32 + k] =
            (k < HD) ? (_Float16)(xsh[k] * invS) : (_Float16)0.f;
}

// ============================================================================
// 4) Base-node prep: x0n[c] = f16(x0[c]/||x0[c]||), rows padded to 32 halfs
// ============================================================================
__global__ __launch_bounds__(256)
void x0prep_kernel(const float* __restrict__ x0, _Float16* __restrict__ x0n)
{
    int c = blockIdx.x * 256 + threadIdx.x;
    if (c >= N0P) return;
    float inv = 0.f;
    if (c < N0C) {
        float s = 0.f;
#pragma unroll
        for (int j = 0; j < HD; ++j) { float v = x0[(size_t)c * HD + j]; s += v * v; }
        inv = rsqrtf(fmaxf(s, 1e-30f));
    }
#pragma unroll
    for (int k = 0; k < 32; ++k)
        x0n[(size_t)c * 32 + k] = (c < N0C && k < HD)
            ? (_Float16)(x0[(size_t)c * HD + k] * inv) : (_Float16)0.f;
}

// ============================================================================
// 5) Fused cosine-sim WMMA + per-row top-6. x0n tiles DMA'd to LDS with
//    global_load_async_to_lds_b128 (double-buffered); sim never materialized.
// ============================================================================
__global__ __launch_bounds__(32)
void topk_kernel(const _Float16* __restrict__ xhn, const _Float16* __restrict__ x0n,
                 int* __restrict__ top5)
{
    __shared__ __align__(16) _Float16 stage[2][16][32];
    __shared__ float simS[16][17];

    const int lane = threadIdx.x;
    const int col  = lane & 15;
    const int hs   = lane >> 4;
    const int m0   = blockIdx.x * 16;

    // A fragment (pre-normalized rows): one contiguous 32B global load, k = hs*16+s
    const _Float16* xr = xhn + (size_t)(m0 + col) * 32;
    v16h a0 = *(const v16h*)(xr + hs * 16);

    const unsigned ldsS[2] = { lds_off(&stage[0][0][0]), lds_off(&stage[1][0][0]) };
    // one tile = 16 rows x 64B = 1024B contiguous in both global and LDS
    auto issue_tile = [&](int nt, unsigned ldsbase) {
        uint64_t gbase = (uint64_t)(uintptr_t)(x0n + (size_t)nt * 16 * 32);
#pragma unroll
        for (int q = 0; q < 2; ++q) {
            unsigned voff = (unsigned)(lane + 32 * q) * 16u;
            asm volatile("global_load_async_to_lds_b128 %0, %1, %2"
                         :: "v"(ldsbase + voff), "v"(voff), "s"(gbase) : "memory");
        }
    };
    issue_tile(0, ldsS[0]);

    float tv[6]; int ti[6];
#pragma unroll
    for (int q = 0; q < 6; ++q) { tv[q] = -1e30f; ti[q] = 0; }

    int buf = 0;
    for (int nt = 0; nt < NT0; ++nt) {
        asm volatile("s_wait_asynccnt 0" ::: "memory");
        __syncthreads();
        if (nt + 1 < NT0) issue_tile(nt + 1, ldsS[buf ^ 1]);

        v16h b = *(const v16h*)&stage[buf][col][hs * 16];
        v8f acc = {};
        acc = __builtin_amdgcn_wmma_f32_16x16x32_f16(false, a0, false, b,
                                                     (short)0, acc, false, false);
        const int c = nt * 16 + col;
#pragma unroll
        for (int r = 0; r < 8; ++r)
            simS[hs * 8 + r][col] = (c < N0C) ? acc[r] : -1e30f;
        __syncthreads();
        if (lane < 16) {
            for (int cc = 0; cc < 16; ++cc) {
                float v = simS[lane][cc];
                int idx = nt * 16 + cc;
                if (v > tv[5]) {
                    tv[5] = v; ti[5] = idx;
#pragma unroll
                    for (int q = 5; q > 0; --q) {
                        if (tv[q] > tv[q - 1]) {
                            float tf = tv[q]; tv[q] = tv[q - 1]; tv[q - 1] = tf;
                            int tx = ti[q]; ti[q] = ti[q - 1]; ti[q - 1] = tx;
                        }
                    }
                }
            }
        }
        __syncthreads();
        buf ^= 1;
    }
    if (lane < 16) {
        bool exact = (tv[0] == 1.0f);  // self-match -> drop first neighbor
        int sel[5];
#pragma unroll
        for (int q = 0; q < 5; ++q) sel[q] = exact ? ti[q + 1] : ti[q];
#pragma unroll
        for (int a = 0; a < 5; ++a)
#pragma unroll
            for (int bq = 0; bq < 4; ++bq)
                if (bq < 4 - a && sel[bq] > sel[bq + 1]) {
                    int t = sel[bq]; sel[bq] = sel[bq + 1]; sel[bq + 1] = t;
                }
#pragma unroll
        for (int q = 0; q < 5; ++q) top5[(size_t)(m0 + lane) * 5 + q] = sel[q];
    }
}

// ============================================================================
// 6) SAGE layer 1: scatter-mean over edge_0, then o1 = relu(Wl mean + Wr x + b1)
// ============================================================================
__global__ __launch_bounds__(256)
void zero_kernel(float* __restrict__ agg, float* __restrict__ cnt)
{
    int i = blockIdx.x * 256 + threadIdx.x;
    if (i < N0C * HD) agg[i] = 0.f;
    if (i < N0C)      cnt[i] = 0.f;
}

__global__ __launch_bounds__(256)
void scatter_kernel(const int* __restrict__ e0, const float* __restrict__ x0,
                    float* __restrict__ agg, float* __restrict__ cnt)
{
    int e = blockIdx.x * 256 + threadIdx.x;
    if (e >= E0C) return;
    int s = e0[e];           // edge_0[0][e]
    int d = e0[E0C + e];     // edge_0[1][e]
#pragma unroll
    for (int j = 0; j < HD; ++j)
        atomicAdd(&agg[(size_t)d * HD + j], x0[(size_t)s * HD + j]);
    atomicAdd(&cnt[d], 1.f);
}

__global__ __launch_bounds__(256)
void sage1_kernel(const float* __restrict__ agg, const float* __restrict__ cnt,
                  const float* __restrict__ x0, const float* __restrict__ xh,
                  const float* __restrict__ w1l, const float* __restrict__ w1r,
                  const float* __restrict__ b1, float* __restrict__ o1)
{
    int i = blockIdx.x * 256 + threadIdx.x;
    if (i >= NTOT * HCC) return;
    int node = i / HCC, h = i % HCC;
    float acc = b1[h];
    if (node < N0C) {
        float invc = 1.f / fmaxf(cnt[node], 1.f);
#pragma unroll
        for (int j = 0; j < HD; ++j)
            acc += w1l[h * HD + j] * agg[(size_t)node * HD + j] * invc;
#pragma unroll
        for (int j = 0; j < HD; ++j)
            acc += w1r[h * HD + j] * x0[(size_t)node * HD + j];
    } else {
        // new nodes receive no edge_0 messages -> mean = 0
#pragma unroll
        for (int j = 0; j < HD; ++j)
            acc += w1r[h * HD + j] * xh[(size_t)(node - N0C) * HD + j];
    }
    o1[i] = fmaxf(acc, 0.f);
}

// ============================================================================
// 7) SAGE layer 2 (new nodes only; exactly their 5 selected base neighbors)
//    + linear head + softmax.
// ============================================================================
__global__ __launch_bounds__(256)
void sage2_kernel(const float* __restrict__ o1, const int* __restrict__ top5,
                  const float* __restrict__ w2l, const float* __restrict__ w2r,
                  const float* __restrict__ b2, const float* __restrict__ wlin,
                  const float* __restrict__ blin, float* __restrict__ out)
{
    int n = blockIdx.x * 256 + threadIdx.x;
    if (n >= NB) return;
    float mean[HCC];
#pragma unroll
    for (int c = 0; c < HCC; ++c) mean[c] = 0.f;
#pragma unroll
    for (int q = 0; q < 5; ++q) {
        int nb = top5[(size_t)n * 5 + q];
#pragma unroll
        for (int c = 0; c < HCC; ++c) mean[c] += o1[(size_t)nb * HCC + c];
    }
#pragma unroll
    for (int c = 0; c < HCC; ++c) mean[c] *= 0.2f;
    const float* self = &o1[(size_t)(N0C + n) * HCC];
    float o2[HD];
#pragma unroll
    for (int h = 0; h < HD; ++h) {
        float acc = b2[h];
#pragma unroll
        for (int c = 0; c < HCC; ++c)
            acc += w2l[h * HCC + c] * mean[c] + w2r[h * HCC + c] * self[c];
        o2[h] = acc;
    }
    float lg[NCLS];
#pragma unroll
    for (int cl = 0; cl < NCLS; ++cl) {
        float acc = blin[cl];
#pragma unroll
        for (int h = 0; h < HD; ++h) acc += wlin[cl * HD + h] * o2[h];
        lg[cl] = acc;
    }
    float mx = fmaxf(lg[0], fmaxf(lg[1], lg[2]));
    float s = 0.f;
#pragma unroll
    for (int cl = 0; cl < NCLS; ++cl) { lg[cl] = __expf(lg[cl] - mx); s += lg[cl]; }
    float inv = 1.f / s;
#pragma unroll
    for (int cl = 0; cl < NCLS; ++cl) out[(size_t)n * NCLS + cl] = lg[cl] * inv;
}

// ============================================================================
extern "C" void kernel_launch(void* const* d_in, const int* in_sizes, int n_in,
                              void* d_out, int out_size, void* d_ws, size_t ws_size,
                              hipStream_t stream)
{
    const float* x    = (const float*)d_in[0];
    const float* x0   = (const float*)d_in[1];
    const float* emb  = (const float*)d_in[2];
    const float* Wih  = (const float*)d_in[3];
    const float* Whh  = (const float*)d_in[4];
    const float* bih  = (const float*)d_in[5];
    const float* bhh  = (const float*)d_in[6];
    const float* aw1  = (const float*)d_in[7];
    const float* ab1  = (const float*)d_in[8];
    const float* aw2  = (const float*)d_in[9];
    const float* ab2  = (const float*)d_in[10];
    const float* w1l  = (const float*)d_in[11];
    const float* w1r  = (const float*)d_in[12];
    const float* b1   = (const float*)d_in[13];
    const float* w2l  = (const float*)d_in[14];
    const float* w2r  = (const float*)d_in[15];
    const float* b2   = (const float*)d_in[16];
    const float* wlin = (const float*)d_in[17];
    const float* blin = (const float*)d_in[18];
    const int*   e0   = (const int*)d_in[19];
    float* out = (float*)d_out;

    char* ws = (char*)d_ws;
    size_t off = 0;
    auto take = [&](size_t bytes) -> void* {
        void* p = ws + off;
        off = (off + bytes + 255) & ~(size_t)255;
        return p;
    };
    _Float16* seqA = (_Float16*)take((size_t)TS * NB * HD * 2);
    _Float16* seqB = (_Float16*)take((size_t)TS * NB * HD * 2);
    float*    xh   = (float*)take((size_t)NB * HD * 4);
    _Float16* xhn  = (_Float16*)take((size_t)NB * 32 * 2);
    _Float16* x0n  = (_Float16*)take((size_t)N0P * 32 * 2);
    int*      top5 = (int*)take((size_t)NB * 5 * 4);
    float*    agg  = (float*)take((size_t)N0C * HD * 4);
    float*    cnt  = (float*)take((size_t)N0C * 4);
    float*    o1   = (float*)take((size_t)NTOT * HCC * 4);

    // 1) embedding -> seqA (f16)
    embed_kernel<<<(TS * NB * HD + 255) / 256, 256, 0, stream>>>(x, emb, seqA);
    // 2) 4 LSTM layers, ping-pong; final output lands in seqA
    for (int l = 0; l < 4; ++l) {
        const _Float16* inp = (l & 1) ? seqB : seqA;
        _Float16* outp      = (l & 1) ? seqA : seqB;
        lstm_layer_kernel<<<NB / 16, 32, 0, stream>>>(
            inp, outp,
            Wih + (size_t)l * GD * HD, Whh + (size_t)l * GD * HD,
            bih + (size_t)l * GD,      bhh + (size_t)l * GD);
    }
    // 3) attention -> xh (f32), xhn (normalized f16)
    attn_kernel<<<NB, 64, 0, stream>>>(seqA, aw1, ab1, aw2, ab2, xh, xhn);
    // 4) base-node normalized f16 rows
    x0prep_kernel<<<(N0P + 255) / 256, 256, 0, stream>>>(x0, x0n);
    // 5) fused cosine + top-k
    topk_kernel<<<NB / 16, 32, 0, stream>>>(xhn, x0n, top5);
    // 6) SAGE layer 1
    zero_kernel<<<(N0C * HD + 255) / 256, 256, 0, stream>>>(agg, cnt);
    scatter_kernel<<<(E0C + 255) / 256, 256, 0, stream>>>(e0, x0, agg, cnt);
    sage1_kernel<<<(NTOT * HCC + 255) / 256, 256, 0, stream>>>(
        agg, cnt, x0, xh, w1l, w1r, b1, o1);
    // 7) SAGE layer 2 + classifier
    sage2_kernel<<<(NB + 255) / 256, 256, 0, stream>>>(
        o1, top5, w2l, w2r, b2, wlin, blin, out);
}
